// JK_78271484003093
// MI455X (gfx1250) — matrix operand order repaired and checked by reading, not compiled, not run on hardware
//
#include <hip/hip_runtime.h>
#include <cstddef>
#include <cstdint>

// Problem constants
#define A_CONST 20.0f
#define B_SZ 2048
#define T_SZ 512
#define H    256   // nh1 == nh2
#define H3   128

typedef __bf16 bf16_t;
typedef bf16_t v16bf __attribute__((ext_vector_type(16)));
typedef bf16_t v8bf  __attribute__((ext_vector_type(8)));
typedef float  v8f   __attribute__((ext_vector_type(8)));

// ---------------- workspace layout (bytes) ----------------
constexpr size_t WS_M1   = 0;                                   // 512 f32
constexpr size_t WS_S1   = WS_M1   + (size_t)512 * 4;           // 512 f32
constexpr size_t WS_CMIN = WS_S1   + (size_t)512 * 4;           // 512*256 f32
constexpr size_t WS_CMAX = WS_CMIN + (size_t)512 * 256 * 4;     // 512*256 f32
constexpr size_t WS_SCAL = WS_CMAX + (size_t)512 * 256 * 4;     // 512*6 f32
constexpr size_t WS_W2BF = WS_SCAL + (size_t)512 * 6 * 4;       // 256*256 bf16
constexpr size_t WS_YPRE = WS_W2BF + (size_t)65536 * 2;         // 512*2048 f32
constexpr size_t WS_Z2   = WS_YPRE + (size_t)512 * 2048 * 4;    // 512*2048*256 bf16

__device__ __forceinline__ float sigf(float z) {
    return 1.0f / (1.0f + __expf(-z));
}
__device__ __forceinline__ unsigned short f2bf(float f) {
    unsigned u = __float_as_uint(f);
    unsigned r = u + 0x7FFFu + ((u >> 16) & 1u);   // round-nearest-even
    return (unsigned short)(r >> 16);
}
__device__ __forceinline__ float bf2f_lo(unsigned d) {  // low bf16 of dword
    return __uint_as_float(d << 16);
}
__device__ __forceinline__ float bf2f_hi(unsigned d) {  // high bf16 of dword
    return __uint_as_float(d & 0xFFFF0000u);
}

// ---------------- K1: per-t X column min/max -> z1 range ----------------
__global__ void k1_stats(const float* __restrict__ X, const float* __restrict__ W1,
                         const float* __restrict__ b1,
                         float* __restrict__ m1ws, float* __restrict__ s1ws) {
    __shared__ float smn[256], smx[256];
    int t = blockIdx.x, tid = threadIdx.x;
    float mn = 3.4e38f, mx = -3.4e38f;
    for (int b = tid; b < B_SZ; b += 256) {
        float v = X[(size_t)b * T_SZ + t];
        mn = fminf(mn, v); mx = fmaxf(mx, v);
    }
    smn[tid] = mn; smx[tid] = mx; __syncthreads();
    for (int s = 128; s > 0; s >>= 1) {
        if (tid < s) { smn[tid] = fminf(smn[tid], smn[tid+s]); smx[tid] = fmaxf(smx[tid], smx[tid+s]); }
        __syncthreads();
    }
    float mnx = smn[0], mxx = smx[0];
    __syncthreads();
    // z1[b,h] = x_b*w_h + b_h : extrema over b are at minx/maxx per sign(w_h)
    float w = W1[tid], bb = b1[tid];
    float zlo = (w >= 0.f ? mnx * w : mxx * w) + bb;
    float zhi = (w >= 0.f ? mxx * w : mnx * w) + bb;
    smn[tid] = zlo; smx[tid] = zhi; __syncthreads();
    for (int s = 128; s > 0; s >>= 1) {
        if (tid < s) { smn[tid] = fminf(smn[tid], smn[tid+s]); smx[tid] = fmaxf(smx[tid], smx[tid+s]); }
        __syncthreads();
    }
    if (tid == 0) {
        float m1 = smn[0], M1 = smx[0];
        m1ws[t] = m1;
        s1ws[t] = A_CONST / (M1 - m1);
    }
}

// ---------------- K1b: W2 -> bf16 ----------------
__global__ void k1b_w2bf(const float* __restrict__ W2, unsigned short* __restrict__ w2bf) {
    int i = blockIdx.x * 256 + threadIdx.x;
    w2bf[i] = f2bf(W2[i]);
}

// ---------------- K2: per-t GEMM z2 = out1 @ W2^T + b2 (WMMA bf16) ----------------
__global__ void __launch_bounds__(256) k2_gemm(
    const float* __restrict__ X, const float* __restrict__ W1, const float* __restrict__ b1,
    const unsigned short* __restrict__ w2bf, const float* __restrict__ b2,
    const float* __restrict__ m1ws, const float* __restrict__ s1ws,
    unsigned short* __restrict__ z2,
    float* __restrict__ cminws, float* __restrict__ cmaxws) {

    __shared__ unsigned short sA[128][264];   // 128 rows x 256 K (+8 pad), bf16
    __shared__ float sX[B_SZ];

    const int t = blockIdx.x, tid = threadIdx.x;
    const int lane = tid & 31, wave = tid >> 5;
    const int laneHi = lane >> 4, lane16 = lane & 15;

    const float m1 = m1ws[t], s1 = s1ws[t];
    for (int b = tid; b < B_SZ; b += 256) sX[b] = X[(size_t)b * T_SZ + t];

    const int nbase = wave * 32;
    const int col0 = nbase + lane16;
    const int col1 = nbase + 16 + lane16;
    const float b2c0 = b2[col0], b2c1 = b2[col1];
    float cmin0 = 3.4e38f, cmax0 = -3.4e38f, cmin1 = 3.4e38f, cmax1 = -3.4e38f;

    const v8f vzero = {0.f, 0.f, 0.f, 0.f, 0.f, 0.f, 0.f, 0.f};

    for (int mb = 0; mb < 16; ++mb) {
        __syncthreads();
        // synthesize out1 tile (128 rows x 256 K) into LDS as bf16
        {
            int r  = tid >> 1;
            int kb = (tid & 1) * 128;
            float xv = sX[mb * 128 + r];
            unsigned* dst = reinterpret_cast<unsigned*>(&sA[r][kb]);
            #pragma unroll 8
            for (int kk = 0; kk < 128; kk += 2) {
                int k = kb + kk;
                float o0 = sigf((xv * W1[k]     + b1[k]     - m1) * s1);
                float o1 = sigf((xv * W1[k + 1] + b1[k + 1] - m1) * s1);
                dst[kk >> 1] = (unsigned)f2bf(o0) | ((unsigned)f2bf(o1) << 16);
            }
        }
        __syncthreads();

        for (int half = 0; half < 2; ++half) {
            v8f acc[4][2];
            #pragma unroll
            for (int i = 0; i < 4; ++i) { acc[i][0] = vzero; acc[i][1] = vzero; }

            #pragma unroll
            for (int k0 = 0; k0 < 256; k0 += 32) {
                // B fragments: lane = N column, 16 contiguous K per lane half
                int kb_ = k0 + laneHi * 16;
                v16bf bf0 = *reinterpret_cast<const v16bf*>(w2bf + (size_t)col0 * H + kb_);
                v16bf bf1 = *reinterpret_cast<const v16bf*>(w2bf + (size_t)col1 * H + kb_);
                #pragma unroll
                for (int mt = 0; mt < 4; ++mt) {
                    int row = half * 64 + mt * 16 + lane16;
                    const unsigned short* ap = &sA[row][0];
                    int kA0 = k0 + laneHi * 16;
                    v8bf lo = *reinterpret_cast<const v8bf*>(ap + kA0);
                    v8bf hi = *reinterpret_cast<const v8bf*>(ap + kA0 + 8);
                    v16bf a = __builtin_shufflevector(lo, hi,
                        0,1,2,3,4,5,6,7,8,9,10,11,12,13,14,15);
                    acc[mt][0] = __builtin_amdgcn_wmma_f32_16x16x32_bf16(
                        false, a, false, bf0, (short)0, acc[mt][0], false, false);
                    acc[mt][1] = __builtin_amdgcn_wmma_f32_16x16x32_bf16(
                        false, a, false, bf1, (short)0, acc[mt][1], false, false);
                }
            }

            // epilogue: +b2, per-column min/max, store z2 (bf16)
            #pragma unroll
            for (int mt = 0; mt < 4; ++mt) {
                int rowg0 = mb * 128 + half * 64 + mt * 16 + laneHi * 8;
                #pragma unroll
                for (int r = 0; r < 8; ++r) {
                    float z0 = acc[mt][0][r] + b2c0;
                    float z1v = acc[mt][1][r] + b2c1;
                    cmin0 = fminf(cmin0, z0);  cmax0 = fmaxf(cmax0, z0);
                    cmin1 = fminf(cmin1, z1v); cmax1 = fmaxf(cmax1, z1v);
                    size_t base = ((size_t)t * B_SZ + rowg0 + r) * H;
                    z2[base + col0] = f2bf(z0);
                    z2[base + col1] = f2bf(z1v);
                }
            }
        }
    }

    // fold the two lane halves (rows 0-7 vs 8-15 of each tile)
    cmin0 = fminf(cmin0, __shfl_xor(cmin0, 16));
    cmax0 = fmaxf(cmax0, __shfl_xor(cmax0, 16));
    cmin1 = fminf(cmin1, __shfl_xor(cmin1, 16));
    cmax1 = fmaxf(cmax1, __shfl_xor(cmax1, 16));
    if (laneHi == 0) {
        cminws[(size_t)t * H + col0] = cmin0;
        cmaxws[(size_t)t * H + col0] = cmax0;
        cminws[(size_t)t * H + col1] = cmin1;
        cmaxws[(size_t)t * H + col1] = cmax1;
    }
}

// ---------------- K3: per-t normalization scalars ----------------
__global__ void k3_scal(const float* __restrict__ cminws, const float* __restrict__ cmaxws,
                        const float* __restrict__ jw, const float* __restrict__ kw,
                        float* __restrict__ scal) {
    __shared__ float smn[256], smx[256], pmn[256], pmx[256];
    int t = blockIdx.x, h = threadIdx.x;
    float cmn = cminws[(size_t)t * H + h];
    float cmx = cmaxws[(size_t)t * H + h];
    smn[h] = cmn; smx[h] = cmx; __syncthreads();
    for (int s = 128; s > 0; s >>= 1) {
        if (h < s) { smn[h] = fminf(smn[h], smn[h+s]); smx[h] = fmaxf(smx[h], smx[h+s]); }
        __syncthreads();
    }
    float gmin = smn[0], gmax = smx[0];
    __syncthreads();
    float s2 = A_CONST / (gmax - gmin);
    // sigmoid is monotone -> column extrema of out2 come from z2 column extrema
    float omn = sigf((cmn - gmin) * s2);
    float omx = sigf((cmx - gmin) * s2);
    float w = (h < H3) ? jw[h] : kw[h - H3];
    pmn[h] = (w >= 0.f) ? omn * w : omx * w;
    pmx[h] = (w >= 0.f) ? omx * w : omn * w;
    __syncthreads();
    if (h == 0) {
        float mnj = 3.4e38f, mxj = -3.4e38f, mnk = 3.4e38f, mxk = -3.4e38f;
        for (int i = 0; i < H3; ++i) { mnj = fminf(mnj, pmn[i]); mxj = fmaxf(mxj, pmx[i]); }
        for (int i = H3; i < H; ++i) { mnk = fminf(mnk, pmn[i]); mxk = fmaxf(mxk, pmx[i]); }
        float* sc = scal + (size_t)t * 6;
        sc[0] = gmin; sc[1] = s2;
        sc[2] = mnj;  sc[3] = A_CONST / (mxj - mnj);
        sc[4] = mnk;  sc[5] = A_CONST / (mxk - mnk);
    }
}

// ---------------- K4: sequential recurrence over t (pipelined loads) ----------------
__global__ void __launch_bounds__(256) k4_scan(
    const unsigned short* __restrict__ z2, const float* __restrict__ scal,
    const float* __restrict__ W4, const float* __restrict__ b4,
    const float* __restrict__ jw, const float* __restrict__ kw,
    float* __restrict__ ypre) {

    const int tid = threadIdx.x;
    const int b   = blockIdx.x * 32 + (tid >> 3);
    const int sub = tid & 7;
    const int h0  = sub * 16;

    float st[16], jv[16], kv[16], w4v[16];
    #pragma unroll
    for (int i = 0; i < 16; ++i) {
        st[i] = 0.f;
        jv[i] = jw[h0 + i];
        kv[i] = kw[h0 + i];
        w4v[i] = W4[h0 + i];
    }
    const float b4v = b4[0];

    const size_t rowoff  = (size_t)b * H;
    const size_t tstride = (size_t)B_SZ * H;

    unsigned cj[8], ck[8];
    {
        const uint4* pj = reinterpret_cast<const uint4*>(z2 + rowoff + h0);
        const uint4* pk = reinterpret_cast<const uint4*>(z2 + rowoff + H3 + h0);
        uint4 a = pj[0], c = pj[1], d = pk[0], e = pk[1];
        cj[0]=a.x; cj[1]=a.y; cj[2]=a.z; cj[3]=a.w; cj[4]=c.x; cj[5]=c.y; cj[6]=c.z; cj[7]=c.w;
        ck[0]=d.x; ck[1]=d.y; ck[2]=d.z; ck[3]=d.w; ck[4]=e.x; ck[5]=e.y; ck[6]=e.z; ck[7]=e.w;
    }

    for (int t = 0; t < T_SZ; ++t) {
        unsigned nj[8], nk[8];
        if (t + 1 < T_SZ) {
            size_t base = (size_t)(t + 1) * tstride + rowoff;
            const uint4* pj = reinterpret_cast<const uint4*>(z2 + base + h0);
            const uint4* pk = reinterpret_cast<const uint4*>(z2 + base + H3 + h0);
            uint4 a = pj[0], c = pj[1], d = pk[0], e = pk[1];
            nj[0]=a.x; nj[1]=a.y; nj[2]=a.z; nj[3]=a.w; nj[4]=c.x; nj[5]=c.y; nj[6]=c.z; nj[7]=c.w;
            nk[0]=d.x; nk[1]=d.y; nk[2]=d.z; nk[3]=d.w; nk[4]=e.x; nk[5]=e.y; nk[6]=e.z; nk[7]=e.w;
            if (t + 2 < T_SZ)
                __builtin_prefetch(z2 + base + tstride + h0, 0, 1);
        }

        const float* sc = scal + (size_t)t * 6;
        float g = sc[0], s2 = sc[1], mj = sc[2], sj = sc[3], mk = sc[4], sk = sc[5];

        float dot = 0.f;
        #pragma unroll
        for (int q = 0; q < 8; ++q) {
            float zj0 = bf2f_lo(cj[q]), zj1 = bf2f_hi(cj[q]);
            float zk0 = bf2f_lo(ck[q]), zk1 = bf2f_hi(ck[q]);
            float ij0 = sigf((sigf((zj0 - g) * s2) * jv[2*q]   - mj) * sj);
            float ij1 = sigf((sigf((zj1 - g) * s2) * jv[2*q+1] - mj) * sj);
            float ik0 = sigf((sigf((zk0 - g) * s2) * kv[2*q]   - mk) * sk);
            float ik1 = sigf((sigf((zk1 - g) * s2) * kv[2*q+1] - mk) * sk);
            st[2*q]   = (1.f - st[2*q])   * ij0 + (1.f - ik0) * st[2*q];
            st[2*q+1] = (1.f - st[2*q+1]) * ij1 + (1.f - ik1) * st[2*q+1];
            dot += st[2*q] * w4v[2*q] + st[2*q+1] * w4v[2*q+1];
        }
        dot += __shfl_xor(dot, 1);
        dot += __shfl_xor(dot, 2);
        dot += __shfl_xor(dot, 4);
        if (sub == 0) ypre[(size_t)t * B_SZ + b] = dot + b4v;

        #pragma unroll
        for (int q = 0; q < 8; ++q) { cj[q] = nj[q]; ck[q] = nk[q]; }
    }
}

// ---------------- K5: per-t min/max over B + final sigmoid ----------------
__global__ void k5_out(const float* __restrict__ ypre, float* __restrict__ out) {
    __shared__ float smn[256], smx[256];
    int t = blockIdx.x, tid = threadIdx.x;
    float v[8];
    float mn = 3.4e38f, mx = -3.4e38f;
    #pragma unroll
    for (int i = 0; i < 8; ++i) {
        v[i] = ypre[(size_t)t * B_SZ + tid + i * 256];
        mn = fminf(mn, v[i]); mx = fmaxf(mx, v[i]);
    }
    smn[tid] = mn; smx[tid] = mx; __syncthreads();
    for (int s = 128; s > 0; s >>= 1) {
        if (tid < s) { smn[tid] = fminf(smn[tid], smn[tid+s]); smx[tid] = fmaxf(smx[tid], smx[tid+s]); }
        __syncthreads();
    }
    float gmn = smn[0], gmx = smx[0];
    float s = A_CONST / (gmx - gmn);
    #pragma unroll
    for (int i = 0; i < 8; ++i) {
        out[(size_t)(tid + i * 256) * T_SZ + t] = sigf((v[i] - gmn) * s);
    }
}

// ---------------- host launch ----------------
extern "C" void kernel_launch(void* const* d_in, const int* in_sizes, int n_in,
                              void* d_out, int out_size, void* d_ws, size_t ws_size,
                              hipStream_t stream) {
    const float* X  = (const float*)d_in[0];
    const float* W1 = (const float*)d_in[1];
    const float* b1 = (const float*)d_in[2];
    const float* W2 = (const float*)d_in[3];
    const float* b2 = (const float*)d_in[4];
    const float* W4 = (const float*)d_in[5];
    const float* b4 = (const float*)d_in[6];
    const float* jw = (const float*)d_in[7];
    const float* kw = (const float*)d_in[8];

    char* ws = (char*)d_ws;
    float*          m1   = (float*)(ws + WS_M1);
    float*          s1   = (float*)(ws + WS_S1);
    float*          cmin = (float*)(ws + WS_CMIN);
    float*          cmax = (float*)(ws + WS_CMAX);
    float*          scal = (float*)(ws + WS_SCAL);
    unsigned short* w2bf = (unsigned short*)(ws + WS_W2BF);
    float*          ypre = (float*)(ws + WS_YPRE);
    unsigned short* z2   = (unsigned short*)(ws + WS_Z2);

    k1_stats<<<T_SZ, 256, 0, stream>>>(X, W1, b1, m1, s1);
    k1b_w2bf<<<256, 256, 0, stream>>>(W2, w2bf);
    k2_gemm<<<T_SZ, 256, 0, stream>>>(X, W1, b1, w2bf, b2, m1, s1, z2, cmin, cmax);
    k3_scal<<<T_SZ, 256, 0, stream>>>(cmin, cmax, jw, kw, scal);
    k4_scan<<<B_SZ / 32, 256, 0, stream>>>(z2, scal, W4, b4, jw, kw, ypre);
    k5_out<<<T_SZ, 256, 0, stream>>>(ypre, (float*)d_out);
}
